// LRR_78056735637975
// MI455X (gfx1250) — matrix operand order repaired and checked by reading, not compiled, not run on hardware
//
#include <hip/hip_runtime.h>
#include <hip/hip_bf16.h>

typedef __attribute__((ext_vector_type(16))) _Float16 v16h;
typedef __attribute__((ext_vector_type(8)))  float    v8f;
typedef __attribute__((ext_vector_type(4)))  _Float16 v4h;

static constexpr int NUM_USER = 8192;
static constexpr int NUM_ITEM = 16384;
static constexpr int KDIM     = 2048;
static constexpr int BATCH    = 2048;
static constexpr float LAMD0 = 40000.0f;
static constexpr float CC    = 5000.0f;

__device__ __forceinline__ float sigmoidf_(float x) { return 1.0f / (1.0f + expf(-x)); }

// ---------------------------------------------------------------------------
// 1) ss[k] = s / (s + LAMD0 + CC*tanh(lamd_w))
// ---------------------------------------------------------------------------
__global__ void k_ss(const float* __restrict__ s, const float* __restrict__ lamd_w,
                     float* __restrict__ ss) {
  int k = blockIdx.x * blockDim.x + threadIdx.x;
  if (k < KDIM) ss[k] = s[k] / (s[k] + LAMD0 + CC * tanhf(lamd_w[k]));
}

// ---------------------------------------------------------------------------
// 2) u (f32) -> u16 (f16), 4 elements per thread
// ---------------------------------------------------------------------------
__global__ void k_u16(const float* __restrict__ u, _Float16* __restrict__ u16) {
  long i = (long)blockIdx.x * blockDim.x + threadIdx.x;  // float4 index
  float4 v = ((const float4*)u)[i];
  v4h h;
  h.x = (_Float16)v.x; h.y = (_Float16)v.y; h.z = (_Float16)v.z; h.w = (_Float16)v.w;
  ((v4h*)u16)[i] = h;
}

// ---------------------------------------------------------------------------
// 3) Ag16[b,i] = UI[user_b, i] * drop[b, i] * sigmoid(D_w[i]); Ag16[b, pos_b] = 0
// ---------------------------------------------------------------------------
__global__ void k_ag16(const float* __restrict__ UI, const float* __restrict__ drop,
                       const float* __restrict__ D_w, const int* __restrict__ user,
                       const int* __restrict__ pos, _Float16* __restrict__ Ag) {
  const int b = blockIdx.x;
  const long row = (long)user[b] * NUM_ITEM;
  const int p = pos[b];
  for (int i = threadIdx.x; i < NUM_ITEM; i += blockDim.x) {
    float v = UI[row + i] * drop[(long)b * NUM_ITEM + i] * sigmoidf_(D_w[i]);
    if (i == p) v = 0.0f;
    Ag[(long)b * NUM_ITEM + i] = (_Float16)v;
  }
}

// ---------------------------------------------------------------------------
// 4) Y[2048,2048] = Ag16[2048,16384] @ u16[16384,2048]  (f16 in, f32 accumulate)
//    128x128 tile per WG, 8 waves (2x4), K staged 64 wide through LDS.
//    - threads 0..127 stage A rows, threads 128..255 stage B with an in-register
//      8x8 transpose so ALL LDS traffic is ds_store_b128 / ds_load_b128.
//    - next K-stage global loads are software-pipelined through registers so
//      HBM latency overlaps the 128 WMMAs of the current stage.
// ---------------------------------------------------------------------------
__global__ __launch_bounds__(256)
void k_gemm(const _Float16* __restrict__ Ag, const _Float16* __restrict__ U,
            float* __restrict__ Y) {
  __shared__ _Float16 As[128 * 80];   // [m][k], row stride 80 halves (160B, 16B aligned)
  __shared__ _Float16 Bs[128 * 80];   // [n][k] (transposed), row stride 80 halves

  const int tid  = threadIdx.x;
  const int lane = tid & 31;
  const int wid  = tid >> 5;
  const int wm   = wid >> 2;          // 0..1 -> 64 rows each
  const int wn   = wid & 3;           // 0..3 -> 32 cols each
  const int m0   = blockIdx.y * 128;
  const int n0   = blockIdx.x * 128;

  v8f acc[4][2];
#pragma unroll
  for (int i = 0; i < 4; ++i)
#pragma unroll
    for (int j = 0; j < 2; ++j) { v8f z = {}; acc[i][j] = z; }

  const int hi = lane >> 4;           // half-wave select
  const int lm = lane & 15;

  // staging roles
  const bool isA = (tid < 128);
  const int  bt  = tid & 127;
  const int  kb8 = (bt >> 4) * 8;     // B: k sub-block base (0..56)
  const int  nb8 = (bt & 15) * 8;     // B: n sub-block base (0..120)

  union Q { uint4 q; _Float16 h[8]; };
  uint4 ra[4];                        // A stager: 32 halves of one A row
  Q     rb[8];                        // B stager: 8x8 block (k-major in)

  // ---- preload stage r0 = 0 into registers ----
  if (isA) {
    const uint4* ga = (const uint4*)(Ag + (long)(m0 + bt) * NUM_ITEM);
#pragma unroll
    for (int j = 0; j < 4; ++j) ra[j] = ga[j];
  } else {
#pragma unroll
    for (int r = 0; r < 8; ++r)
      rb[r].q = *(const uint4*)(U + (long)(kb8 + r) * KDIM + n0 + nb8);
  }

  for (int r0 = 0; r0 < NUM_ITEM; r0 += 64) {
    // ---- commit registers -> LDS ----
    if (isA) {
      uint4* sa = (uint4*)(&As[bt * 80]);
#pragma unroll
      for (int j = 0; j < 4; ++j) sa[j] = ra[j];
    } else {
#pragma unroll
      for (int e = 0; e < 8; ++e) {
        Q t;
#pragma unroll
        for (int r = 0; r < 8; ++r) t.h[r] = rb[r].h[e];   // transpose in regs
        *(uint4*)(&Bs[(nb8 + e) * 80 + kb8]) = t.q;
      }
    }
    __syncthreads();

    // ---- issue next stage's global loads (overlap with WMMA below) ----
    const int rn = r0 + 64;
    if (rn < NUM_ITEM) {
      if (isA) {
        const uint4* ga = (const uint4*)(Ag + (long)(m0 + bt) * NUM_ITEM + rn);
#pragma unroll
        for (int j = 0; j < 4; ++j) ra[j] = ga[j];
      } else {
#pragma unroll
        for (int r = 0; r < 8; ++r)
          rb[r].q = *(const uint4*)(U + (long)(rn + kb8 + r) * KDIM + n0 + nb8);
      }
    }

    // ---- compute on LDS tiles ----
#pragma unroll
    for (int kk = 0; kk < 64; kk += 32) {
      union Frag { v16h v; uint4 q[2]; };
      Frag a[4], b[2];
      // A frag (16x32 f16): halves 0..7 -> K=kk+hi*8.., 8..15 -> K=kk+16+hi*8..
#pragma unroll
      for (int mt = 0; mt < 4; ++mt) {
        const int M = wm * 64 + mt * 16 + lm;
        a[mt].q[0] = *(const uint4*)&As[M * 80 + kk + hi * 8];
        a[mt].q[1] = *(const uint4*)&As[M * 80 + kk + 16 + hi * 8];
      }
      // B frag (32x16 f16): lane N=lm, halves 0..7 -> K=kk+hi*8.., 8..15 -> K=kk+16+hi*8..
#pragma unroll
      for (int nt = 0; nt < 2; ++nt) {
        const int N = wn * 32 + nt * 16 + lm;
        b[nt].q[0] = *(const uint4*)&Bs[N * 80 + kk + hi * 8];
        b[nt].q[1] = *(const uint4*)&Bs[N * 80 + kk + 16 + hi * 8];
      }
#pragma unroll
      for (int mt = 0; mt < 4; ++mt)
#pragma unroll
        for (int nt = 0; nt < 2; ++nt)
          acc[mt][nt] = __builtin_amdgcn_wmma_f32_16x16x32_f16(
              false, a[mt].v, false, b[nt].v, (short)0, acc[mt][nt], false, false);
    }
    __syncthreads();
  }

  // ---- write C: lane holds N=lm (hi selects M+8), VGPR r -> row ----
#pragma unroll
  for (int mt = 0; mt < 4; ++mt)
#pragma unroll
    for (int nt = 0; nt < 2; ++nt) {
      const int N = n0 + wn * 32 + nt * 16 + lm;
#pragma unroll
      for (int r = 0; r < 8; ++r) {
        const int M = m0 + wm * 64 + mt * 16 + hi * 8 + r;
        Y[(long)M * KDIM + N] = acc[mt][nt][r];
      }
    }
}

// ---------------------------------------------------------------------------
// 5) finalize: per-b dots + gates + diag correction + stable -log_sigmoid
// ---------------------------------------------------------------------------
__global__ void k_zero(float* out) { out[0] = 0.0f; }

__global__ __launch_bounds__(256)
void k_final(const float* __restrict__ Y, const float* __restrict__ ss,
             const float* __restrict__ vh, const float* __restrict__ u,
             const float* __restrict__ UI, const float* __restrict__ drop,
             const float* __restrict__ D_w, const float* __restrict__ Db_w,
             const int* __restrict__ user, const int* __restrict__ pos,
             const int* __restrict__ neg, float* __restrict__ out) {
  __shared__ float s0[256], s1[256], s2[256];
  const int b = blockIdx.x, tid = threadIdx.x;
  const int p = pos[b], n = neg[b], us = user[b];
  float dp = 0.0f, dn = 0.0f, du = 0.0f;
  for (int k = tid; k < KDIM; k += 256) {
    float y  = Y[(long)b * KDIM + k];
    float sk = ss[k];
    float zp = sk * vh[(long)k * NUM_ITEM + p];
    float zn = sk * vh[(long)k * NUM_ITEM + n];
    dp += y * zp;
    dn += y * zn;
    du += u[(long)n * KDIM + k] * zn;
  }
  s0[tid] = dp; s1[tid] = dn; s2[tid] = du;
  __syncthreads();
  for (int st = 128; st > 0; st >>= 1) {
    if (tid < st) { s0[tid] += s0[tid + st]; s1[tid] += s1[tid + st]; s2[tid] += s2[tid + st]; }
    __syncthreads();
  }
  if (tid == 0) {
    float dbgp = sigmoidf_(Db_w[p]);
    float dbgn = sigmoidf_(Db_w[n]);
    float posv = dbgp * s0[0];
    float Agbn = (n == p) ? 0.0f
                          : UI[(long)us * NUM_ITEM + n] * drop[(long)b * NUM_ITEM + n] *
                            sigmoidf_(D_w[n]);
    float negv = dbgn * (s1[0] - Agbn * s2[0]);
    float v = 100.0f * (posv - negv);
    float li = (v >= 0.0f) ? log1pf(expf(-v)) : (-v + log1pf(expf(v)));
    atomicAdd(out, li * (1.0f / (float)BATCH));
  }
}

// ---------------------------------------------------------------------------
extern "C" void kernel_launch(void* const* d_in, const int* in_sizes, int n_in,
                              void* d_out, int out_size, void* d_ws, size_t ws_size,
                              hipStream_t stream) {
  const float* UI     = (const float*)d_in[0];
  const float* u      = (const float*)d_in[1];
  const float* s      = (const float*)d_in[2];
  const float* vh     = (const float*)d_in[3];
  const float* lamd_w = (const float*)d_in[4];
  const float* D_w    = (const float*)d_in[5];
  const float* Db_w   = (const float*)d_in[6];
  const float* drop   = (const float*)d_in[7];
  const int*   user   = (const int*)d_in[8];
  const int*   pos    = (const int*)d_in[9];
  const int*   neg    = (const int*)d_in[10];
  float* out = (float*)d_out;

  char* ws = (char*)d_ws;
  _Float16* u16  = (_Float16*)ws;                       // 16384*2048*2 = 64 MiB
  _Float16* Ag16 = (_Float16*)(ws + 67108864);          // 2048*16384*2 = 64 MiB
  float*    Y    = (float*)(ws + 134217728);            // 2048*2048*4  = 16 MiB
  float*    ssb  = (float*)(ws + 150994944);            // 2048*4

  k_ss<<<KDIM / 256, 256, 0, stream>>>(s, lamd_w, ssb);
  k_u16<<<(int)(((long)NUM_ITEM * KDIM / 4) / 256), 256, 0, stream>>>(u, u16);
  k_ag16<<<BATCH, 256, 0, stream>>>(UI, drop, D_w, user, pos, Ag16);
  dim3 grid(KDIM / 128, BATCH / 128);
  k_gemm<<<grid, 256, 0, stream>>>(Ag16, u16, Y);
  k_zero<<<1, 1, 0, stream>>>(out);
  k_final<<<BATCH, 256, 0, stream>>>(Y, ssb, vh, u, UI, drop, D_w, Db_w, user, pos, neg, out);
}
